// MultiHeadSelfAttention_7954279432240
// MI455X (gfx1250) — matrix operand order repaired and checked by reading, not compiled, not run on hardware
//
#include <hip/hip_runtime.h>

// ---------------------------------------------------------------------------
// MI455X (gfx1250) fused MHA forward:
//   cvt fp32->bf16 -> 3x GEMM (QKV, bf16 out) -> flash attention ->
//   GEMM (output projection, fp32 out)
// GEMMs run on v_wmma_f32_16x16x32_bf16; tile staging uses the Tensor Data
// Mover (tensor_load_to_lds + s_wait_tensorcnt) with DMA-side LDS padding;
// softmax row reductions use v_permlane16_b32 (no LDS pipe).
// ---------------------------------------------------------------------------

typedef __attribute__((ext_vector_type(16))) __bf16 v16bf;
typedef __attribute__((ext_vector_type(8)))  float  v8f;
typedef __attribute__((ext_vector_type(4)))  unsigned int v4u;
typedef __attribute__((ext_vector_type(8)))  int v8i;
typedef __attribute__((ext_vector_type(4)))  int v4i;

union FragBF {
  v16bf v;
  uint4 q[2];
  unsigned short u[16];
};

__device__ __forceinline__ unsigned short f2bf(float f) {
  unsigned int u = __float_as_uint(f);
  unsigned int lsb = (u >> 16) & 1u;              // round-to-nearest-even
  return (unsigned short)((u + 0x7FFFu + lsb) >> 16);
}

// A-fragment (16x32, 16-bit): lane holds row m=lane%16; VGPR0-3 = K half*8..+7,
// VGPR4-7 = K 16+half*8..+7  (two contiguous 16B chunks per lane).
__device__ __forceinline__ v16bf load_a_frag(const unsigned short* base, int ld) {
  int lane = threadIdx.x & 31;
  int m = lane & 15, half = lane >> 4;
  const unsigned short* p = base + m * ld + half * 8;
  FragBF f;
  f.q[0] = *(const uint4*)(p);
  f.q[1] = *(const uint4*)(p + 16);
  return f.v;
}

// B-fragment (32x16, 16-bit): lane holds column n=lane%16; VGPR0-3 = K half*8..+7,
// VGPR4-7 = K 16+half*8..+7  (two contiguous 16B chunks per lane).
__device__ __forceinline__ v16bf load_b_frag(const unsigned short* base, int ld) {
  int lane = threadIdx.x & 31;
  int n = lane & 15, half = lane >> 4;
  const unsigned short* p = base + n * ld + half * 8;
  FragBF f;
  f.q[0] = *(const uint4*)(p);
  f.q[1] = *(const uint4*)(p + 16);
  return f.v;
}

__device__ __forceinline__ void store_out(float* C, size_t idx, float v) { C[idx] = v; }
__device__ __forceinline__ void store_out(unsigned short* C, size_t idx, float v) { C[idx] = f2bf(v); }

// v_permlane16_b32 xor-pattern shuffle within each 16-lane half (pure VALU).
__device__ __forceinline__ float permxor16(float v, int s1, int s2) {
  int r = __builtin_amdgcn_permlane16(__float_as_int(v), __float_as_int(v), s1, s2,
                                      /*fi=*/true, /*bound_ctrl=*/false);
  return __int_as_float(r);
}
// nibble tables: dest lane i reads lane i^m (same table applies in both rows)
#define XOR1_LO 0x67452301
#define XOR1_HI ((int)0xEFCDAB89u)
#define XOR2_LO 0x54761032
#define XOR2_HI ((int)0xDCFE98BAu)
#define XOR4_LO 0x32107654
#define XOR4_HI ((int)0xBA98FEDCu)
#define XOR8_LO ((int)0xFEDCBA98u)
#define XOR8_HI 0x76543210

// ---------------------------------------------------------------------------
// Tensor Data Mover: DMA a (tile_h x tile_w) bf16 tile (row-major, row stride
// `stride_elems` in global) into LDS. pad_enable inserts 4 DWORDs (16B) after
// every 32 DWORDs (one 128B row of 64 bf16), producing the 144B (LDT=72)
// padded LDS row stride the WMMA fragment loaders expect.  D# per ISA §8.
// Issue from ONE wave only (EXEC is ignored; op is wave-level).
// ---------------------------------------------------------------------------
__device__ __forceinline__ void tdm_load_tile_bf16(const unsigned short* gsrc,
                                                   unsigned short* lds_dst,
                                                   unsigned stride_elems,
                                                   unsigned tile_w,   // 64
                                                   unsigned tile_h) {
  unsigned long long ga = (unsigned long long)gsrc;
  unsigned lds_addr = (unsigned)(unsigned long long)lds_dst;  // LDS aperture: low 32 bits
  v4u g0;
  g0[0] = 1u;                                   // count=1, user descriptor
  g0[1] = lds_addr;                             // lds_addr [63:32]
  g0[2] = (unsigned)ga;                         // global_addr lo
  g0[3] = (unsigned)(ga >> 32) | (2u << 30);    // global_addr hi | type=2
  const unsigned data_size = 1;                 // 2-byte elements
  const unsigned pad_en = 1, pad_interval = 4;  // pad after 32 DWORDs (128B)
  const unsigned pad_amount = 3;                //  ... by 4 DWORDs (16B)
  unsigned td0 = stride_elems;                  // tensor_dim0 = full row width
  unsigned td1 = 0x40000000u;                   // tensor_dim1: large (no clip)
  v8i g1;
  g1[0] = (int)((data_size << 16) | (pad_en << 20) | (pad_interval << 22) |
                (pad_amount << 25));            // wg_mask=0
  g1[1] = (int)((td0 & 0xFFFFu) << 16);         // tensor_dim0[15:0]
  g1[2] = (int)((td0 >> 16) | ((td1 & 0xFFFFu) << 16));
  g1[3] = (int)((td1 >> 16) | (tile_w << 16)); // tile_dim0
  g1[4] = (int)(tile_h);                        // tile_dim1 | tile_dim2=0
  g1[5] = (int)stride_elems;                    // tensor_dim0_stride[31:0]
  g1[6] = 0;                                    // stride0 hi | stride1 lo
  g1[7] = 0;
  v4i g2 = {0, 0, 0, 0};
  v4i g3 = {0, 0, 0, 0};
#if defined(__clang_major__) && __clang_major__ >= 23
  v8i g4 = {0, 0, 0, 0, 0, 0, 0, 0};
  __builtin_amdgcn_tensor_load_to_lds(g0, g1, g2, g3, g4, 0);
#else
  __builtin_amdgcn_tensor_load_to_lds(g0, g1, g2, g3, 0);
#endif
}

// ---------------------------------------------------------------------------
// fp32 -> bf16 conversion, float4 vectorized
// ---------------------------------------------------------------------------
__global__ __launch_bounds__(256) void cvt_f32_to_bf16(const float* __restrict__ in,
                                                       unsigned short* __restrict__ out,
                                                       int n4) {
  int i = blockIdx.x * blockDim.x + threadIdx.x;
  if (i >= n4) return;
  float4 f = ((const float4*)in)[i];
  ushort4 o;
  o.x = f2bf(f.x); o.y = f2bf(f.y); o.z = f2bf(f.z); o.w = f2bf(f.w);
  ((ushort4*)out)[i] = o;
}

// ---------------------------------------------------------------------------
// C[m,n] = sum_k A[m,k]*B[n,k] + bias[n]    (x @ W^T + b)
// Block 128x128, K-step 64, 8 waves (2x4), wave computes 64x32.
// Double-buffered LDS; tiles arrive via TDM DMA issued by wave 0 so the
// next tile's HBM traffic overlaps the current tile's 16 WMMAs.
// ---------------------------------------------------------------------------
template <typename OutT>
__global__ __launch_bounds__(256) void gemm_xwT_bias(
    const unsigned short* __restrict__ A,   // M x K bf16
    const unsigned short* __restrict__ Bw,  // N x K bf16
    const float* __restrict__ bias,         // N
    OutT* __restrict__ C,                   // M x N
    int M, int N, int K) {
  constexpr int BM = 128, BN = 128, BK = 64, LDT = 72;
  __shared__ unsigned short sA[2][BM * LDT];
  __shared__ unsigned short sB[2][BN * LDT];
  int tid = threadIdx.x;
  int lane = tid & 31, wid = tid >> 5;
  int wm = wid & 1, wn = wid >> 1;            // 2 (M) x 4 (N) wave grid
  int m0 = blockIdx.x * BM, n0 = blockIdx.y * BN;

  v8f acc[4][2];
#pragma unroll
  for (int i = 0; i < 4; ++i)
#pragma unroll
    for (int j = 0; j < 2; ++j) acc[i][j] = {};

  int nTiles = K / BK;
  if (wid == 0) {  // prologue: DMA tile 0
    tdm_load_tile_bf16(&A[(size_t)m0 * K], &sA[0][0], (unsigned)K, BK, BM);
    tdm_load_tile_bf16(&Bw[(size_t)n0 * K], &sB[0][0], (unsigned)K, BK, BN);
  }
  for (int t = 0; t < nTiles; ++t) {
    __syncthreads();  // all waves done with the buffer tile t+1 overwrites
    if (wid == 0) {
      if (t + 1 < nTiles) {
        int kt = (t + 1) * BK, nb = (t + 1) & 1;
        tdm_load_tile_bf16(&A[(size_t)m0 * K + kt], &sA[nb][0], (unsigned)K, BK, BM);
        tdm_load_tile_bf16(&Bw[(size_t)n0 * K + kt], &sB[nb][0], (unsigned)K, BK, BN);
        __builtin_amdgcn_s_wait_tensorcnt(2);   // tile t complete (in-order)
      } else {
        __builtin_amdgcn_s_wait_tensorcnt(0);
      }
    }
    __syncthreads();  // tile t visible to the workgroup
    const unsigned short* cA = &sA[t & 1][0];
    const unsigned short* cB = &sB[t & 1][0];
#pragma unroll
    for (int kc = 0; kc < BK; kc += 32) {
      v16bf af[4], bfr[2];
#pragma unroll
      for (int mi = 0; mi < 4; ++mi)
        af[mi] = load_a_frag(&cA[(wm * 64 + mi * 16) * LDT + kc], LDT);
#pragma unroll
      for (int nj = 0; nj < 2; ++nj)
        bfr[nj] = load_b_frag(&cB[(wn * 32 + nj * 16) * LDT + kc], LDT);
#pragma unroll
      for (int mi = 0; mi < 4; ++mi)
#pragma unroll
        for (int nj = 0; nj < 2; ++nj)
          acc[mi][nj] = __builtin_amdgcn_wmma_f32_16x16x32_bf16(
              false, af[mi], false, bfr[nj], (short)0, acc[mi][nj], false, false);
    }
  }

  // epilogue: C-layout lane (half=lane/16, n=lane%16), VGPR j -> row j+8*half
  int n16 = lane & 15, half = lane >> 4;
#pragma unroll
  for (int nj = 0; nj < 2; ++nj) {
    int gc = n0 + wn * 32 + nj * 16 + n16;
    float bv = bias[gc];
#pragma unroll
    for (int mi = 0; mi < 4; ++mi)
#pragma unroll
      for (int j = 0; j < 8; ++j) {
        int gr = m0 + wm * 64 + mi * 16 + j + 8 * half;
        store_out(C, (size_t)gr * N + gc, acc[mi][nj][j] + bv);
      }
  }
}

// ---------------------------------------------------------------------------
// Flash attention: 256 threads (8 waves) per block = 128 queries of one
// (batch, head). Wave owns 16 query rows; Q frags in registers; online
// softmax with permlane16 reductions; K/Q tiles via TDM, V transposed
// manually (overlaps the K DMA); P transposed C->A layout through LDS.
// ---------------------------------------------------------------------------
__global__ __launch_bounds__(256) void mha_flash(
    const unsigned short* __restrict__ Qg,
    const unsigned short* __restrict__ Kg,
    const unsigned short* __restrict__ Vg,
    unsigned short* __restrict__ Ctx) {
  constexpr int LDT = 72, T = 2048, NU = 1024;
  __shared__ unsigned short sK[64 * LDT];     // K tile, row-major [t][d]
  __shared__ unsigned short sVT[64 * LDT];    // V^T tile [d][t]
  __shared__ unsigned short sQP[128 * LDT];   // Q staging, then per-wave P tiles
  const float scale = 0.125f;                 // 1/sqrt(64)

  int tid = threadIdx.x, lane = tid & 31, wid = tid >> 5;
  int n16 = lane & 15, half = lane >> 4;
  int qBlk = blockIdx.x * 128;
  int b = blockIdx.y >> 4, h = blockIdx.y & 15;
  size_t rowBase = (size_t)b * T;
  int colBase = h * 64;

  // stage Q tile (128 x 64) via TDM
  if (wid == 0) {
    tdm_load_tile_bf16(&Qg[(rowBase + qBlk) * NU + colBase], &sQP[0], NU, 64, 128);
    __builtin_amdgcn_s_wait_tensorcnt(0);
  }
  __syncthreads();

  int qr0 = wid * 16;                         // this wave's private 16 rows
  v16bf qf0 = load_a_frag(&sQP[qr0 * LDT + 0], LDT);
  v16bf qf1 = load_a_frag(&sQP[qr0 * LDT + 32], LDT);

  float mrow[8], lrow[8];
  v8f acc[4];
#pragma unroll
  for (int j = 0; j < 8; ++j) { mrow[j] = -3.0e30f; lrow[j] = 0.0f; }
#pragma unroll
  for (int dt = 0; dt < 4; ++dt) acc[dt] = {};

  for (int kb = 0; kb < T; kb += 64) {
    __syncthreads();  // previous tile fully consumed
    if (wid == 0)     // K tile DMA overlaps the manual V transpose below
      tdm_load_tile_bf16(&Kg[(rowBase + kb) * NU + colBase], &sK[0], NU, 64, 64);
    {  // stage V^T: each thread moves 16 bf16, transposing into [d][t]
      int r = tid >> 2, c = (tid & 3) * 16;
      FragBF vv;
      const uint4* vp = (const uint4*)&Vg[(rowBase + kb + r) * NU + colBase + c];
      vv.q[0] = vp[0];
      vv.q[1] = vp[1];
#pragma unroll
      for (int e = 0; e < 16; ++e) sVT[(c + e) * LDT + r] = vv.u[e];
    }
    if (wid == 0) __builtin_amdgcn_s_wait_tensorcnt(0);
    __syncthreads();

    // S = Q K^T : 4 key sub-tiles of 16, contraction D=64 in two K32 steps
    v8f s[4];
#pragma unroll
    for (int nt = 0; nt < 4; ++nt) {
      v16bf b0 = load_b_frag(&sK[(nt * 16) * LDT + 0], LDT);
      v16bf b1 = load_b_frag(&sK[(nt * 16) * LDT + 32], LDT);
      v8f sv = {};
      sv = __builtin_amdgcn_wmma_f32_16x16x32_bf16(false, qf0, false, b0, (short)0, sv, false, false);
      sv = __builtin_amdgcn_wmma_f32_16x16x32_bf16(false, qf1, false, b1, (short)0, sv, false, false);
      s[nt] = sv;
    }

    // online softmax: lane holds 8 rows (j+8*half) of one column n16;
    // row reductions across the 16-lane halves via v_permlane16_b32.
    float mx[8];
#pragma unroll
    for (int j = 0; j < 8; ++j) mx[j] = -3.0e30f;
#pragma unroll
    for (int nt = 0; nt < 4; ++nt)
#pragma unroll
      for (int j = 0; j < 8; ++j) {
        s[nt][j] *= scale;
        mx[j] = fmaxf(mx[j], s[nt][j]);
      }
#pragma unroll
    for (int j = 0; j < 8; ++j) {
      mx[j] = fmaxf(mx[j], permxor16(mx[j], XOR1_LO, XOR1_HI));
      mx[j] = fmaxf(mx[j], permxor16(mx[j], XOR2_LO, XOR2_HI));
      mx[j] = fmaxf(mx[j], permxor16(mx[j], XOR4_LO, XOR4_HI));
      mx[j] = fmaxf(mx[j], permxor16(mx[j], XOR8_LO, XOR8_HI));
    }

    float alpha[8], rs[8];
#pragma unroll
    for (int j = 0; j < 8; ++j) {
      float mn = fmaxf(mrow[j], mx[j]);
      alpha[j] = __expf(mrow[j] - mn);
      mrow[j] = mn;
      rs[j] = 0.0f;
    }
#pragma unroll
    for (int nt = 0; nt < 4; ++nt)
#pragma unroll
      for (int j = 0; j < 8; ++j) {
        float p = __expf(s[nt][j] - mrow[j]);
        s[nt][j] = p;
        rs[j] += p;
      }
#pragma unroll
    for (int j = 0; j < 8; ++j) {
      rs[j] += permxor16(rs[j], XOR1_LO, XOR1_HI);
      rs[j] += permxor16(rs[j], XOR2_LO, XOR2_HI);
      rs[j] += permxor16(rs[j], XOR4_LO, XOR4_HI);
      rs[j] += permxor16(rs[j], XOR8_LO, XOR8_HI);
    }
#pragma unroll
    for (int j = 0; j < 8; ++j) lrow[j] = lrow[j] * alpha[j] + rs[j];
#pragma unroll
    for (int dt = 0; dt < 4; ++dt)
#pragma unroll
      for (int j = 0; j < 8; ++j) acc[dt][j] *= alpha[j];

    // P (C-layout) -> per-wave LDS rows [qr0, qr0+16) as bf16, row-major
#pragma unroll
    for (int nt = 0; nt < 4; ++nt)
#pragma unroll
      for (int j = 0; j < 8; ++j)
        sQP[(qr0 + j + 8 * half) * LDT + nt * 16 + n16] = f2bf(s[nt][j]);
    // same-wave DS ops are in-order; wait for the stores, fence the compiler
    asm volatile("s_wait_dscnt 0" ::: "memory");
    __builtin_amdgcn_wave_barrier();

    // ctx += P @ V : A = P (16x64), B = V^T tile (column d, contiguous t)
    v16bf pf0 = load_a_frag(&sQP[qr0 * LDT + 0], LDT);
    v16bf pf1 = load_a_frag(&sQP[qr0 * LDT + 32], LDT);
#pragma unroll
    for (int dt = 0; dt < 4; ++dt) {
      v16bf vb0 = load_b_frag(&sVT[(dt * 16) * LDT + 0], LDT);
      v16bf vb1 = load_b_frag(&sVT[(dt * 16) * LDT + 32], LDT);
      acc[dt] = __builtin_amdgcn_wmma_f32_16x16x32_bf16(false, pf0, false, vb0, (short)0, acc[dt], false, false);
      acc[dt] = __builtin_amdgcn_wmma_f32_16x16x32_bf16(false, pf1, false, vb1, (short)0, acc[dt], false, false);
    }
  }

  // normalize and emit bf16 ctx (B*T x 1024 layout, head-sliced columns)
#pragma unroll
  for (int j = 0; j < 8; ++j) {
    float inv = 1.0f / lrow[j];
    size_t gr = rowBase + qBlk + qr0 + j + 8 * half;
#pragma unroll
    for (int dt = 0; dt < 4; ++dt)
      Ctx[gr * NU + colBase + dt * 16 + n16] = f2bf(acc[dt][j] * inv);
  }
}

// ---------------------------------------------------------------------------
extern "C" void kernel_launch(void* const* d_in, const int* in_sizes, int n_in,
                              void* d_out, int out_size, void* d_ws, size_t ws_size,
                              hipStream_t stream) {
  const float* x  = (const float*)d_in[0];
  const float* Wq = (const float*)d_in[1];
  const float* bq = (const float*)d_in[2];
  const float* Wk = (const float*)d_in[3];
  const float* bk = (const float*)d_in[4];
  const float* Wv = (const float*)d_in[5];
  const float* bv = (const float*)d_in[6];
  const float* Wo = (const float*)d_in[7];
  const float* bo = (const float*)d_in[8];
  float* out = (float*)d_out;

  const int Mrows = 8192, NU = 1024;    // B*T, n_units

  // workspace layout (bf16): x, 4 weights, Q, K, V, ctx  (~92 MB total)
  unsigned short* w = (unsigned short*)d_ws;
  unsigned short* xb  = w; w += (size_t)Mrows * NU;
  unsigned short* wqb = w; w += (size_t)NU * NU;
  unsigned short* wkb = w; w += (size_t)NU * NU;
  unsigned short* wvb = w; w += (size_t)NU * NU;
  unsigned short* wob = w; w += (size_t)NU * NU;
  unsigned short* qb  = w; w += (size_t)Mrows * NU;
  unsigned short* kb  = w; w += (size_t)Mrows * NU;
  unsigned short* vb  = w; w += (size_t)Mrows * NU;
  unsigned short* ctxb = w;

  {
    int n4 = Mrows * NU / 4;
    cvt_f32_to_bf16<<<(n4 + 255) / 256, 256, 0, stream>>>(x, xb, n4);
    n4 = NU * NU / 4;
    cvt_f32_to_bf16<<<(n4 + 255) / 256, 256, 0, stream>>>(Wq, wqb, n4);
    cvt_f32_to_bf16<<<(n4 + 255) / 256, 256, 0, stream>>>(Wk, wkb, n4);
    cvt_f32_to_bf16<<<(n4 + 255) / 256, 256, 0, stream>>>(Wv, wvb, n4);
    cvt_f32_to_bf16<<<(n4 + 255) / 256, 256, 0, stream>>>(Wo, wob, n4);
  }

  dim3 gg(Mrows / 128, NU / 128);
  gemm_xwT_bias<unsigned short><<<gg, 256, 0, stream>>>(xb, wqb, bq, qb, Mrows, NU, NU);
  gemm_xwT_bias<unsigned short><<<gg, 256, 0, stream>>>(xb, wkb, bk, kb, Mrows, NU, NU);
  gemm_xwT_bias<unsigned short><<<gg, 256, 0, stream>>>(xb, wvb, bv, vb, Mrows, NU, NU);

  dim3 ga(2048 / 128, 4 * 16);          // (T/128, B*H)
  mha_flash<<<ga, 256, 0, stream>>>(qb, kb, vb, ctxb);

  gemm_xwT_bias<float><<<gg, 256, 0, stream>>>(ctxb, wob, bo, out, Mrows, NU, NU);
}
